// Nets_71554155151852
// MI455X (gfx1250) — compile-verified, matrix operand order checked by reading, not run on hardware
//
#include <hip/hip_runtime.h>

// ---------------------------------------------------------------------------
// Problem constants (match the reference)
// ---------------------------------------------------------------------------
#define NN    8000
#define KK    8
#define EE    (NN*KK)          // 64000 edges
#define CC    32
#define SS    9
#define KCS   (CC*SS)          // 288 = 9 * 32 (exact WMMA K-steps)
#define FF    128
#define BB    128
#define HH    4
#define COUT  64

typedef __bf16 bf16;
typedef __attribute__((ext_vector_type(16))) __bf16 v16bf;
typedef __attribute__((ext_vector_type(8)))  __bf16 v8bf;
typedef __attribute__((ext_vector_type(8)))  float  v8f;

// Gaussian expansion: centers = linspace(0, 6, 128); width = 0.5*6/128
#define GE_STEP   (6.0f / 127.0f)
#define GE_INV2W2 (16384.0f / 18.0f)

// Swizzled-weight element counts (bf16):
//   per matrix block layout: (((kk*ntiles)+nt)*32 + lane)*16 + j
#define W2_ELEMS    (9 * 8 * 512)    // 36864  (288x128)
#define WR_ELEMS    (4 * 8 * 512)    // 16384  (128x128)
#define WAIN_ELEMS  (4 * 4 * 512)    //  8192  (128x64, per branch)
#define WAMID_ELEMS (2 * 4 * 512)    //  4096  (64x64,  per branch)
#define WLIN_ELEMS  (4 * 4 * 512)    //  8192  (128x64)

// ---------------------------------------------------------------------------
// WMMA helpers (bf16 in, f32 accumulate), layouts per CDNA5 ISA 7.12.2
// ---------------------------------------------------------------------------
__device__ __forceinline__ v8f wmma_bf16(v16bf a, v16bf b, v8f c) {
  return __builtin_amdgcn_wmma_f32_16x16x32_bf16(
      /*neg_a=*/false, a, /*neg_b=*/false, b,
      /*c_mod=*/(short)0, c, /*reuse_a=*/false, /*reuse_b=*/false);
}

// A fragment: 16x32 bf16 from row-major [m][k] (LDS).  Two contiguous
// 16-byte runs per lane -> two ds_load_b128.
// lane<16: row m=lane,   K = k0+0..7  and  k0+16..23
// lane>=16: row m=lane-16, K offset +8.
__device__ __forceinline__ v16bf load_a_frag(const bf16* __restrict__ A,
                                             int lda, int lane, int k0) {
  int m    = lane & 15;
  int koff = (lane >> 4) ? 8 : 0;
  const bf16* base = A + m * lda + k0 + koff;
  v8bf lo = *(const v8bf*)(base);
  v8bf hi = *(const v8bf*)(base + 16);
  return __builtin_shufflevector(lo, hi, 0, 1, 2, 3, 4, 5, 6, 7,
                                 8, 9, 10, 11, 12, 13, 14, 15);
}

// B fragment from pre-swizzled weights: one aligned 32-byte load per lane.
__device__ __forceinline__ v16bf load_b_swz(const bf16* __restrict__ Bsw,
                                            int ntiles, int kk, int nt,
                                            int lane) {
  return *(const v16bf*)(Bsw + (((size_t)(kk * ntiles + nt) * 32 + lane) << 4));
}

// Tiled GEMM: C[MT*16][NT*16] (f32, LDS) = A(bf16, row-major) * Bsw(bf16,
// fragment-swizzled).  K = KSTEPS*32.  8 waves; one 16x16 tile per wave/iter.
template <int KSTEPS, int MT, int NT>
__device__ __forceinline__ void gemm_tiles(const bf16* __restrict__ A, int lda,
                                           const bf16* __restrict__ Bsw,
                                           float* __restrict__ Cout, int ldc,
                                           int wave, int lane) {
  constexpr int total = MT * NT;
  for (int tile = wave; tile < total; tile += 8) {
    int mt = tile / NT, nt = tile % NT;
    v8f acc = {};
#pragma unroll
    for (int kk = 0; kk < KSTEPS; ++kk) {
      v16bf a = load_a_frag(A + mt * 16 * lda, lda, lane, kk * 32);
      v16bf b = load_b_swz(Bsw, NT, kk, nt, lane);
      acc = wmma_bf16(a, b, acc);
    }
    int n = lane & 15, mh = (lane >> 4) << 3;
#pragma unroll
    for (int r = 0; r < 8; ++r)
      Cout[(mt * 16 + r + mh) * ldc + nt * 16 + n] = acc[r];
  }
}

// ---------------------------------------------------------------------------
// Kernel 0: convert + swizzle weights into WMMA B-fragment order.
// Source layout: row-major [K][N].  Dest: (((kk*ntiles)+nt)*32+lane)*16+j with
//   n = lane&15, koff = (lane>=16)?8:0, k = kk*32 + koff + (j&7) + (j>=8?16:0)
// ---------------------------------------------------------------------------
__device__ __forceinline__ int swz_src(int idx, int ntiles) {
  int j    = idx & 15;
  int lane = (idx >> 4) & 31;
  int blk  = idx >> 9;            // kk*ntiles + nt
  int nt   = blk % ntiles;
  int kk   = blk / ntiles;
  int n    = lane & 15;
  int koff = (lane >> 4) ? 8 : 0;
  int k    = kk * 32 + koff + (j & 7) + ((j >= 8) ? 16 : 0);
  return k * (ntiles * 16) + nt * 16 + n;
}

__global__ __launch_bounds__(256) void k_convert(
    const float* __restrict__ W_tp2, const float* __restrict__ W_rad,
    const float* __restrict__ Wa_in, const float* __restrict__ Wa_mid,
    const float* __restrict__ W_lin,
    bf16* __restrict__ W2sw, bf16* __restrict__ Wrsw,
    bf16* __restrict__ Wainsw, bf16* __restrict__ Wamidsw,
    bf16* __restrict__ Wlinsw) {
  int idx = blockIdx.x * 256 + threadIdx.x;
  if (idx < W2_ELEMS) W2sw[idx] = (bf16)W_tp2[swz_src(idx, 8)];
  if (idx < WR_ELEMS) Wrsw[idx] = (bf16)W_rad[swz_src(idx, 8)];
  if (idx < 2 * WAIN_ELEMS) {
    int br = idx / WAIN_ELEMS, r = idx % WAIN_ELEMS;
    Wainsw[idx] = (bf16)Wa_in[br * (BB * 64) + swz_src(r, 4)];
  }
  if (idx < 2 * WAMID_ELEMS) {
    int br = idx / WAMID_ELEMS, r = idx % WAMID_ELEMS;
    Wamidsw[idx] = (bf16)Wa_mid[br * (64 * 64) + swz_src(r, 4)];
  }
  if (idx < WLIN_ELEMS) Wlinsw[idx] = (bf16)W_lin[swz_src(idx, 4)];
}

// ---------------------------------------------------------------------------
// Kernel 1: value[E][128] = (outer(edge_in, edge_sh) @ W_tp2) * (gexp @ W_rad)
// One block = 16 edges (one M tile).  8 waves = 8 N-tiles of F=128.
// ---------------------------------------------------------------------------
__global__ __launch_bounds__(256) void k_value(
    const float* __restrict__ edge_in, const float* __restrict__ edge_sh,
    const float* __restrict__ ele,
    const bf16* __restrict__ W2sw, const bf16* __restrict__ Wrsw,
    float* __restrict__ value) {
  __shared__ float sin_[16][CC];
  __shared__ float ssh[16][SS];
  __shared__ bf16  sA2[16][KCS];   // outer-product rows (K = 288)
  __shared__ bf16  sGE[16][BB];    // length embedding rows (K = 128)

  int e0 = blockIdx.x * 16;
  int tid = threadIdx.x;
  int wave = tid >> 5, lane = tid & 31;

  __builtin_prefetch(W2sw + (size_t)wave * 4096, 0, 3);
  __builtin_prefetch(Wrsw + (size_t)wave * 2048, 0, 3);

  for (int idx = tid; idx < 16 * CC; idx += 256)
    sin_[idx >> 5][idx & 31] = edge_in[(size_t)e0 * CC + idx];
  if (tid < 16 * SS) ssh[tid / SS][tid % SS] = edge_sh[(size_t)e0 * SS + tid];
  for (int idx = tid; idx < 16 * BB; idx += 256) {
    int r = idx >> 7, j = idx & 127;
    sGE[r][j] = (bf16)(ele[(size_t)(e0 + r) * BB + j]);
  }
  __syncthreads();
  for (int idx = tid; idx < 16 * KCS; idx += 256) {
    int r = idx / KCS, cs = idx % KCS;
    int c = cs / SS, s = cs % SS;
    sA2[r][cs] = (bf16)(sin_[r][c] * ssh[r][s]);
  }
  __syncthreads();

  v8f acc = {};
#pragma unroll
  for (int kk = 0; kk < KCS / 32; ++kk) {        // 9 K-steps
    v16bf a = load_a_frag(&sA2[0][0], KCS, lane, kk * 32);
    v16bf b = load_b_swz(W2sw, 8, kk, wave, lane);
    acc = wmma_bf16(a, b, acc);
  }
  v8f accr = {};
#pragma unroll
  for (int kk = 0; kk < BB / 32; ++kk) {         // 4 K-steps
    v16bf a = load_a_frag(&sGE[0][0], BB, lane, kk * 32);
    v16bf b = load_b_swz(Wrsw, 8, kk, wave, lane);
    accr = wmma_bf16(a, b, accr);
  }

  int n = lane & 15, mh = (lane >> 4) << 3;
#pragma unroll
  for (int r = 0; r < 8; ++r) {
    int m = r + mh;
    value[(size_t)(e0 + m) * FF + wave * 16 + n] = acc[r] * accr[r];
  }
}

// ---------------------------------------------------------------------------
// LayerNorm(+bias) + SiLU over one 64-wide row, f32 LDS in -> bf16 LDS out
// ---------------------------------------------------------------------------
__device__ __forceinline__ void ln_silu_row(const float* __restrict__ Hf,
                                            bf16* __restrict__ Hb, int row,
                                            const float* __restrict__ bi,
                                            const float* __restrict__ g,
                                            const float* __restrict__ be) {
  const float* x = Hf + row * 64;
  float mu = 0.f;
  for (int j = 0; j < 64; ++j) mu += x[j] + bi[j];
  mu *= (1.f / 64.f);
  float var = 0.f;
  for (int j = 0; j < 64; ++j) { float d = x[j] + bi[j] - mu; var += d * d; }
  var *= (1.f / 64.f);
  float rs = rsqrtf(var + 1e-6f);
  bf16* o = Hb + row * 64;
  for (int j = 0; j < 64; ++j) {
    float h = (x[j] + bi[j] - mu) * rs * g[j] + be[j];
    o[j] = (bf16)(h / (1.f + __expf(-h)));   // SiLU
  }
}

// ---------------------------------------------------------------------------
// Kernel 2: one workgroup per node.  Fuses triplet geometry, gaussian
// expansion, both alpha MLPs (WMMA), segment softmax (groups of 8), weighted
// value reduction, WMMA 128->64 projection, and node scatter-add.
// Exploits: triplet t = i*64 + a*8 + b; branch-1 MLP depends only on b.
// ---------------------------------------------------------------------------
__global__ __launch_bounds__(256) void k_node(
    const float* __restrict__ edge_vec,   // [E][3]
    const float* __restrict__ value,      // [E][128]  (scratch, f32)
    const bf16*  __restrict__ Wainsw,     // [2] x swizzled 128x64
    const bf16*  __restrict__ Wamidsw,    // [2] x swizzled 64x64
    const bf16*  __restrict__ Wlinsw,     // swizzled 128x64
    const float* __restrict__ ba_in, const float* __restrict__ ga1,
    const float* __restrict__ bea1,
    const float* __restrict__ ba_mid, const float* __restrict__ ga2,
    const float* __restrict__ bea2,
    const float* __restrict__ Wa_out,     // [2][64][4] f32
    const float* __restrict__ ba_out,     // [2][4]     f32
    const int*   __restrict__ inv_index,  // [E]
    const int*   __restrict__ edge_dst,   // [E]
    float* __restrict__ node_out)         // [N][64]
{
  __shared__ float ev[KK][3];
  __shared__ int   inv8[KK];
  __shared__ float vrows[KK][FF];     // value[inv_index[i*8+b]]
  __shared__ float rikn[KK];
  __shared__ float rjkn[64];
  __shared__ bf16  X2[64][BB];        // gexp(|rjk|) per triplet
  __shared__ bf16  X1[16][BB];        // gexp(|rik|) rows / edge_fea rows
  __shared__ float Hf[64][64];        // GEMM output (pre-LN)
  __shared__ bf16  Hb[64][64];        // post-activation (next-layer A)
  __shared__ float a2mat[64][HH];
  __shared__ float a1vec[KK][HH];
  __shared__ float alpha[64][HH];

  int i = blockIdx.x;
  int tid = threadIdx.x;
  int wave = tid >> 5, lane = tid & 31;

  __builtin_prefetch(Wainsw + (size_t)wave * 2048, 0, 3);
  __builtin_prefetch(Wamidsw + (size_t)wave * 1024, 0, 3);

  // ---- stage geometry + gathered value rows --------------------------------
  if (tid < KK * 3) ((float*)ev)[tid] = edge_vec[(size_t)i * KK * 3 + tid];
  if (tid < KK)     inv8[tid] = inv_index[i * KK + tid];
  __syncthreads();
  {
    // 8 rows x 128 f32 = 256 float4 loads, one per thread
    int b = tid >> 5, f4 = tid & 31;
    const float4* src =
        (const float4*)(value + (size_t)inv8[b] * FF) + f4;
    ((float4*)&vrows[b][0])[f4] = *src;
  }
  if (tid < KK) {
    float x = ev[tid][0], y = ev[tid][1], z = ev[tid][2];
    rikn[tid] = sqrtf(x * x + y * y + z * z);
  }
  if (tid < 64) {
    int a = tid >> 3, b = tid & 7;
    float dx = ev[b][0] - ev[a][0];
    float dy = ev[b][1] - ev[a][1];
    float dz = ev[b][2] - ev[a][2];
    rjkn[tid] = sqrtf(dx * dx + dy * dy + dz * dz);
  }
  __syncthreads();

  // ---- gaussian expansions --------------------------------------------------
  for (int idx = tid; idx < 64 * BB; idx += 256) {
    int t = idx >> 7, j = idx & 127;
    float d = rjkn[t] - GE_STEP * (float)j;
    X2[t][j] = (bf16)__expf(-d * d * GE_INV2W2);
  }
  for (int idx = tid; idx < 16 * BB; idx += 256) {
    int t = idx >> 7, j = idx & 127;
    float v = 0.f;
    if (t < KK) {
      float d = rikn[t] - GE_STEP * (float)j;
      v = __expf(-d * d * GE_INV2W2);
    }
    X1[t][j] = (bf16)v;
  }
  __syncthreads();

  // ---- branch 2 (rjk) MLP: 64 rows ----------------------------------------
  gemm_tiles<4, 4, 4>(&X2[0][0], BB, Wainsw + WAIN_ELEMS, &Hf[0][0], 64, wave, lane);
  __syncthreads();
  if (tid < 64) ln_silu_row(&Hf[0][0], &Hb[0][0], tid, ba_in + 64, ga1 + 64, bea1 + 64);
  __syncthreads();
  gemm_tiles<2, 4, 4>(&Hb[0][0], 64, Wamidsw + WAMID_ELEMS, &Hf[0][0], 64, wave, lane);
  __syncthreads();
  if (tid < 64) ln_silu_row(&Hf[0][0], &Hb[0][0], tid, ba_mid + 64, ga2 + 64, bea2 + 64);
  __syncthreads();
  if (tid < 64) {
    for (int h = 0; h < HH; ++h) {
      float s = ba_out[HH + h];
      for (int j = 0; j < 64; ++j)
        s += (float)Hb[tid][j] * Wa_out[64 * HH + j * HH + h];
      a2mat[tid][h] = s;
    }
  }
  __syncthreads();

  // ---- branch 1 (rik) MLP: 8 real rows (padded to 16) ----------------------
  gemm_tiles<4, 1, 4>(&X1[0][0], BB, Wainsw, &Hf[0][0], 64, wave, lane);
  __syncthreads();
  if (tid < 16) {
    if (tid < KK) ln_silu_row(&Hf[0][0], &Hb[0][0], tid, ba_in, ga1, bea1);
    else for (int j = 0; j < 64; ++j) Hb[tid][j] = (bf16)0.f;
  }
  __syncthreads();
  gemm_tiles<2, 1, 4>(&Hb[0][0], 64, Wamidsw, &Hf[0][0], 64, wave, lane);
  __syncthreads();
  if (tid < KK) ln_silu_row(&Hf[0][0], &Hb[0][0], tid, ba_mid, ga2, bea2);
  __syncthreads();
  if (tid < KK) {
    for (int h = 0; h < HH; ++h) {
      float s = ba_out[h];
      for (int j = 0; j < 64; ++j)
        s += (float)Hb[tid][j] * Wa_out[j * HH + h];
      a1vec[tid][h] = s;
    }
  }
  __syncthreads();

  // ---- alpha = a1 * a2, segment softmax over b (groups of 8) ----------------
  if (tid < 64) {
    int b = tid & 7;
    for (int h = 0; h < HH; ++h) alpha[tid][h] = a1vec[b][h] * a2mat[tid][h];
  }
  __syncthreads();
  if (tid < 32) {
    int a = tid >> 2, h = tid & 3;
    float mx = -1e30f;
    for (int b = 0; b < KK; ++b) mx = fmaxf(mx, alpha[a * KK + b][h]);
    float s = 0.f;
    for (int b = 0; b < KK; ++b) {
      float e = __expf(alpha[a * KK + b][h] - mx);
      alpha[a * KK + b][h] = e;
      s += e;
    }
    float inv = 1.f / (s + 1e-16f);
    for (int b = 0; b < KK; ++b) alpha[a * KK + b][h] *= inv;
  }
  __syncthreads();

  // ---- edge_fea[a][f] = sum_b alpha * vrows[b][f]  (bf16, into X1, padded) --
  for (int idx = tid; idx < 16 * FF; idx += 256) {
    int a = idx >> 7, f = idx & 127;
    float s = 0.f;
    if (a < KK) {
      int h = f >> 5;
#pragma unroll
      for (int b = 0; b < KK; ++b) s += alpha[a * KK + b][h] * vrows[b][f];
    }
    X1[a][f] = (bf16)s;
  }
  __syncthreads();

  // ---- edge_out = edge_fea @ W_lin via WMMA; scatter-add to nodes -----------
  gemm_tiles<4, 1, 4>(&X1[0][0], BB, Wlinsw, &Hf[0][0], 64, wave, lane);
  __syncthreads();
  for (int idx = tid; idx < KK * COUT; idx += 256) {
    int a = idx >> 6, o = idx & 63;
    int d = edge_dst[i * KK + a];
    __hip_atomic_fetch_add(&node_out[(size_t)d * COUT + o], Hf[a][o],
                           __ATOMIC_RELAXED, __HIP_MEMORY_SCOPE_AGENT);
  }
}

// ---------------------------------------------------------------------------
// Launch
// ---------------------------------------------------------------------------
extern "C" void kernel_launch(void* const* d_in, const int* in_sizes, int n_in,
                              void* d_out, int out_size, void* d_ws, size_t ws_size,
                              hipStream_t stream) {
  const float* edge_in  = (const float*)d_in[0];
  const float* edge_sh  = (const float*)d_in[1];
  const float* ele      = (const float*)d_in[2];
  const float* edge_vec = (const float*)d_in[3];
  const float* W_tp2    = (const float*)d_in[4];
  const float* W_rad    = (const float*)d_in[5];
  const float* W_lin    = (const float*)d_in[6];
  const float* Wa_in    = (const float*)d_in[7];
  const float* ba_in    = (const float*)d_in[8];
  const float* ga1      = (const float*)d_in[9];
  const float* bea1     = (const float*)d_in[10];
  const float* Wa_mid   = (const float*)d_in[11];
  const float* ba_mid   = (const float*)d_in[12];
  const float* ga2      = (const float*)d_in[13];
  const float* bea2     = (const float*)d_in[14];
  const float* Wa_out   = (const float*)d_in[15];
  const float* ba_out   = (const float*)d_in[16];
  const int*   inv_index= (const int*)d_in[17];
  // d_in[18] = tgt_eid, d_in[19] = src_eid: implied by t = i*64 + a*8 + b
  const int*   edge_dst = (const int*)d_in[20];
  float* node_out = (float*)d_out;

  // scratch layout (bytes, all >=1KB aligned)
  char* ws = (char*)d_ws;
  bf16* W2sw    = (bf16*)(ws + 0);          //  73728 B
  bf16* Wrsw    = (bf16*)(ws + 73728);      //  32768 B
  bf16* Wainsw  = (bf16*)(ws + 106496);     //  32768 B
  bf16* Wamidsw = (bf16*)(ws + 139264);     //  16384 B
  bf16* Wlinsw  = (bf16*)(ws + 155648);     //  16384 B
  float* value  = (float*)(ws + 180224);    //  E*128*4 = 32768000 B

  hipMemsetAsync(d_out, 0, (size_t)out_size * sizeof(float), stream);

  k_convert<<<(W2_ELEMS + 255) / 256, 256, 0, stream>>>(
      W_tp2, W_rad, Wa_in, Wa_mid, W_lin, W2sw, Wrsw, Wainsw, Wamidsw, Wlinsw);

  k_value<<<EE / 16, 256, 0, stream>>>(edge_in, edge_sh, ele, W2sw, Wrsw, value);

  k_node<<<NN, 256, 0, stream>>>(edge_vec, value, Wainsw, Wamidsw, Wlinsw,
                                 ba_in, ga1, bea1, ba_mid, ga2, bea2,
                                 Wa_out, ba_out,
                                 inv_index, edge_dst, node_out);
  (void)in_sizes; (void)n_in; (void)ws_size;
}
